// TinyMamba3_26276609917227
// MI455X (gfx1250) — matrix operand order repaired and checked by reading, not compiled, not run on hardware
//
#include <hip/hip_runtime.h>
#include <hip/hip_bf16.h>

// ---------------- model constants ----------------
#define D_MODEL   256
#define D_INNER   512
#define D_STATE   8
#define HEADDIM   16
#define NHEADS    32
#define CHUNK     8
#define D_CONV    4
#define CONV_DIM  528          // D_INNER + 2*D_STATE
#define D_INPROJ  1104         // 2*D_INNER + 2*D_STATE + 2*NHEADS
#define INPUT_DIM 57
#define OUT_SIZE  6
#define BATCH     16
#define SEQLEN    4096
#define M_TOT     (BATCH*SEQLEN)   // 65536 tokens
#define KPAD      64               // INPUT_DIM padded for WMMA K
#define NTILES    (D_INPROJ/16)    // 69
#define MTILES    (M_TOT/16)       // 4096

typedef __attribute__((ext_vector_type(16))) _Float16 v16h;
typedef __attribute__((ext_vector_type(8)))  _Float16 v8h;
typedef __attribute__((ext_vector_type(8)))  float    v8f;

// ---------------- kernel 1: pad+convert x to f16 [M_TOT][64] ----------------
__global__ void k_prep_x(const float* __restrict__ x, _Float16* __restrict__ xh) {
    int gid = blockIdx.x * blockDim.x + threadIdx.x;
    int m = gid >> 6, k = gid & 63;
    float v = (k < INPUT_DIM) ? x[(size_t)m * INPUT_DIM + k] : 0.f;
    xh[gid] = (_Float16)v;
}

// ------- kernel 2: W_comb = W_inproj @ W_in, stored transposed f16 [64][1104];
//                   b_comb = W_inproj @ b_in --------
__global__ void k_wcomb(const float* __restrict__ W_inproj, const float* __restrict__ W_in,
                        const float* __restrict__ b_in,
                        _Float16* __restrict__ wct, float* __restrict__ bcomb) {
    int n = blockIdx.x;          // 0..1103
    int k = threadIdx.x;         // 0..63
    float s = 0.f;
    if (k < INPUT_DIM) {
        for (int d = 0; d < D_MODEL; ++d)
            s += W_inproj[n * D_MODEL + d] * W_in[d * INPUT_DIM + k];
    }
    wct[(size_t)k * D_INPROJ + n] = (_Float16)s;   // transposed: [k][n]
    if (k == 0) {
        float bs = 0.f;
        for (int d = 0; d < D_MODEL; ++d) bs += W_inproj[n * D_MODEL + d] * b_in[d];
        bcomb[n] = bs;
    }
}

// ------- kernel 3: zxbcdt = x_pad @ W_comb^T + b_comb  (WMMA f16, f32 accum) -------
// one wave32 computes one 16x16 tile; K = 64 = 2 steps of 32
__global__ void k_gemm_inproj(const _Float16* __restrict__ xh, const _Float16* __restrict__ wct,
                              const float* __restrict__ bcomb, float* __restrict__ zx) {
    int wave = (blockIdx.x * blockDim.x + threadIdx.x) >> 5;
    int lane = threadIdx.x & 31;
    if (wave >= MTILES * NTILES) return;
    int mt = wave / NTILES, nt = wave % NTILES;
    int m0 = mt * 16, n0 = nt * 16;
    int half = lane >> 4;        // 0 or 1
    int r    = lane & 15;
    v8f acc = {};
#pragma unroll
    for (int ks = 0; ks < 2; ++ks) {
        // A fragment (16x32 f16): lane holds row m0+r; half selects K sub-chunks
        const _Float16* ap = xh + (size_t)(m0 + r) * KPAD + ks * 32 + half * 8;
        v8h alo = *(const v8h*)(ap);        // K = base..base+7
        v8h ahi = *(const v8h*)(ap + 16);   // K = base+16..base+23
        v16h a;
#pragma unroll
        for (int i = 0; i < 8; ++i) { a[i] = alo[i]; a[8 + i] = ahi[i]; }
        // B fragment (32x16 f16): lane holds col n0+r; half selects K sub-chunks (same as A); wct is [K][N]
        const _Float16* bp = wct + (size_t)(ks * 32 + half * 8) * D_INPROJ + n0 + r;
        v16h bfrag;
#pragma unroll
        for (int i = 0; i < 8; ++i) { bfrag[i] = bp[(size_t)i * D_INPROJ]; bfrag[8 + i] = bp[(size_t)(16 + i) * D_INPROJ]; }
        acc = __builtin_amdgcn_wmma_f32_16x16x32_f16(
                  false, a, false, bfrag, (short)0, acc, false, false);
    }
    float bias = bcomb[n0 + r];
    // D layout: VGPR i -> row m0 + half*8 + i, col n0 + r
#pragma unroll
    for (int i = 0; i < 8; ++i)
        zx[(size_t)(m0 + half * 8 + i) * D_INPROJ + n0 + r] = acc[i] + bias;
}

// ------- kernel 4: depthwise causal conv(4) + SiLU over xBC channels -------
__global__ void k_conv(const float* __restrict__ zx, const float* __restrict__ conv_w,
                       const float* __restrict__ conv_b, float* __restrict__ xbca) {
    size_t gid = (size_t)blockIdx.x * blockDim.x + threadIdx.x;
    if (gid >= (size_t)M_TOT * CONV_DIM) return;
    int c = (int)(gid % CONV_DIM);
    int m = (int)(gid / CONV_DIM);
    int t = m & (SEQLEN - 1);         // position within batch sequence
    float acc = conv_b[c];
#pragma unroll
    for (int k = 0; k < D_CONV; ++k) {
        int tt = t - (D_CONV - 1) + k;
        if (tt >= 0)
            acc += zx[(size_t)(m - (D_CONV - 1) + k) * D_INPROJ + D_INNER + c]
                   * conv_w[c * D_CONV + k];
    }
    xbca[gid] = acc / (1.f + __expf(-acc));      // SiLU
}

// ------- kernel 5: dt = softplus(dt_raw + dt_bias) -------
__global__ void k_dt(const float* __restrict__ zx, const float* __restrict__ dt_bias,
                     float* __restrict__ dts) {
    int gid = blockIdx.x * blockDim.x + threadIdx.x;
    int h = gid & (NHEADS - 1);
    int m = gid >> 5;
    float v = zx[(size_t)m * D_INPROJ + (D_INPROJ - NHEADS) + h] + dt_bias[h];
    dts[gid] = (v > 20.f) ? v : log1pf(__expf(v));
}

// ------- kernel 6: chunked SSM scan + gate + pooled-sum reduction -------
// one block (128 threads) per (batch, head); state h[16][8] in LDS; 512 chunks sequential
__global__ void k_ssm(const float* __restrict__ xbca, const float* __restrict__ dts,
                      const float* __restrict__ zx, const float* __restrict__ A_log,
                      const float* __restrict__ Dvec, float* __restrict__ ysum) {
    __shared__ float hst[HEADDIM][D_STATE];
    __shared__ float xc[CHUNK][HEADDIM];
    __shared__ float Bc[CHUNK][D_STATE];
    __shared__ float Cc[CHUNK][D_STATE];
    __shared__ float dtc[CHUNK];
    int t = threadIdx.x;
    int b = blockIdx.x >> 5, h = blockIdx.x & 31;
    float A  = -__expf(A_log[h]);
    float Dh = Dvec[h];
    int i  = t >> 4, p  = t & 15;    // phase-A mapping (row-in-chunk, head dim)
    int p2 = t >> 3, n2 = t & 7;     // phase-B mapping (head dim, state dim)
    hst[p2][n2] = 0.f;
    float accp = 0.f;
    const int NC = SEQLEN / CHUNK;
    for (int c = 0; c < NC; ++c) {
        int m0 = b * SEQLEN + c * CHUNK;
        // cooperative chunk loads
        xc[i][p] = xbca[(size_t)(m0 + i) * CONV_DIM + h * HEADDIM + p];
        if (t < 64)
            Bc[t >> 3][t & 7] = xbca[(size_t)(m0 + (t >> 3)) * CONV_DIM + D_INNER + (t & 7)];
        else
            Cc[(t - 64) >> 3][(t - 64) & 7] =
                xbca[(size_t)(m0 + ((t - 64) >> 3)) * CONV_DIM + D_INNER + D_STATE + ((t - 64) & 7)];
        if (t < CHUNK) dtc[t] = dts[(size_t)(m0 + t) * NHEADS + h];
        __syncthreads();
        // per-thread cumulative dt*A
        float ca[CHUNK];
        { float s = 0.f;
#pragma unroll
          for (int j = 0; j < CHUNK; ++j) { s += dtc[j] * A; ca[j] = s; } }
        // ---- phase A: y = intra + inter + D*x, gated by SiLU(z) ----
        float xi = xc[i][p];
        float y  = Dh * xi;
        float ei = __expf(ca[i]);
        float hsum = 0.f;
#pragma unroll
        for (int n = 0; n < D_STATE; ++n) hsum += hst[p][n] * Cc[i][n];
        y += hsum * ei;                       // inter-chunk via running state
        for (int j = 0; j <= i; ++j) {        // intra-chunk (causal)
            float cb = 0.f;
#pragma unroll
            for (int n = 0; n < D_STATE; ++n) cb += Cc[i][n] * Bc[j][n];
            y += cb * __expf(ca[i] - ca[j]) * dtc[j] * xc[j][p];
        }
        float zz = zx[(size_t)(m0 + i) * D_INPROJ + h * HEADDIM + p];
        y *= zz / (1.f + __expf(-zz));        // gate with SiLU(z)
        accp += y;
        __syncthreads();
        // ---- phase B: state decay + chunk contribution ----
        float caL = ca[CHUNK - 1];
        float s = 0.f;
#pragma unroll
        for (int j = 0; j < CHUNK; ++j)
            s += __expf(caL - ca[j]) * dtc[j] * Bc[j][n2] * xc[j][p2];
        hst[p2][n2] = hst[p2][n2] * __expf(caL) + s;
        __syncthreads();
    }
    // channel = h*16 + p; 8 threads (i=0..7) share a channel -> atomic reduce
    atomicAdd(&ysum[b * D_INNER + h * HEADDIM + p], accp);
}

// ------- kernel 7: pooled = (ysum/L) @ W_out^T  [16,256] -------
__global__ void k_pool(const float* __restrict__ ysum, const float* __restrict__ W_out,
                       float* __restrict__ pooled) {
    int b = blockIdx.x, d = threadIdx.x;
    float s = 0.f;
    for (int k = 0; k < D_INNER; ++k) s += ysum[b * D_INNER + k] * W_out[d * D_INNER + k];
    pooled[b * D_MODEL + d] = s * (1.f / (float)SEQLEN);
}

// ------- kernel 8: out = pooled @ W_cls^T + b_cls  [16,6] -------
__global__ void k_cls(const float* __restrict__ pooled, const float* __restrict__ W_cls,
                      const float* __restrict__ b_cls, float* __restrict__ out) {
    int t = threadIdx.x;
    if (t >= BATCH * OUT_SIZE) return;
    int b = t / OUT_SIZE, o = t % OUT_SIZE;
    float s = b_cls[o];
    for (int d = 0; d < D_MODEL; ++d) s += pooled[b * D_MODEL + d] * W_cls[o * D_MODEL + d];
    out[t] = s;
}

extern "C" void kernel_launch(void* const* d_in, const int* in_sizes, int n_in,
                              void* d_out, int out_size, void* d_ws, size_t ws_size,
                              hipStream_t stream) {
    (void)in_sizes; (void)n_in; (void)out_size; (void)ws_size;
    const float* x        = (const float*)d_in[0];
    const float* W_in     = (const float*)d_in[1];
    const float* b_in     = (const float*)d_in[2];
    const float* W_inproj = (const float*)d_in[3];
    const float* conv_w   = (const float*)d_in[4];
    const float* conv_b   = (const float*)d_in[5];
    const float* dt_bias  = (const float*)d_in[6];
    const float* A_log    = (const float*)d_in[7];
    const float* Dvec     = (const float*)d_in[8];
    const float* W_out    = (const float*)d_in[9];
    const float* W_cls    = (const float*)d_in[10];
    const float* b_cls    = (const float*)d_in[11];
    float* out = (float*)d_out;

    // workspace carve-up (256-B aligned)
    char* ws = (char*)d_ws;
    size_t off = 0;
    auto carve = [&](size_t bytes) {
        void* p = ws + off;
        off = (off + bytes + 255) & ~(size_t)255;
        return p;
    };
    _Float16* xh   = (_Float16*)carve((size_t)M_TOT * KPAD * 2);        //   8 MB
    _Float16* wct  = (_Float16*)carve((size_t)KPAD * D_INPROJ * 2);     // 138 KB
    float*    bcomb= (float*)carve((size_t)D_INPROJ * 4);
    float*    zx   = (float*)carve((size_t)M_TOT * D_INPROJ * 4);       // 289 MB
    float*    xbca = (float*)carve((size_t)M_TOT * CONV_DIM * 4);       // 138 MB
    float*    dts  = (float*)carve((size_t)M_TOT * NHEADS * 4);         //   8 MB
    float*    ysum = (float*)carve((size_t)BATCH * D_INNER * 4);
    float*    pooled=(float*)carve((size_t)BATCH * D_MODEL * 4);

    hipMemsetAsync(ysum, 0, (size_t)BATCH * D_INNER * 4, stream);

    k_prep_x<<<(M_TOT * KPAD) / 256, 256, 0, stream>>>(x, xh);
    k_wcomb<<<D_INPROJ, 64, 0, stream>>>(W_inproj, W_in, b_in, wct, bcomb);

    int tiles = MTILES * NTILES;                 // 282624 tiles, 8 waves per block
    k_gemm_inproj<<<(tiles + 7) / 8, 256, 0, stream>>>(xh, wct, bcomb, zx);

    size_t convN = (size_t)M_TOT * CONV_DIM;
    k_conv<<<(unsigned)((convN + 255) / 256), 256, 0, stream>>>(zx, conv_w, conv_b, xbca);
    k_dt<<<(M_TOT * NHEADS) / 256, 256, 0, stream>>>(zx, dt_bias, dts);

    k_ssm<<<BATCH * NHEADS, 128, 0, stream>>>(xbca, dts, zx, A_log, Dvec, ysum);

    k_pool<<<BATCH, D_MODEL, 0, stream>>>(ysum, W_out, pooled);
    k_cls<<<1, 128, 0, stream>>>(pooled, W_cls, b_cls, out);
}
